// ONNX_TRT2_39333310496773
// MI455X (gfx1250) — matrix operand order repaired and checked by reading, not compiled, not run on hardware
//
#include <hip/hip_runtime.h>
#include <cstddef>

typedef __attribute__((ext_vector_type(16))) _Float16 v16h;
typedef __attribute__((ext_vector_type(8)))  float    v8f;

namespace {
constexpr int kB  = 16;
constexpr int kN  = 25200;
constexpr int kNM = 32;
constexpr int kH  = 160;
constexpr int kW  = 160;
constexpr int kHW = kH * kW;            // 25600
constexpr int kM  = 100;
constexpr int kNC = 80;
constexpr int kXS = 5 + kNC + kNM;      // 117 floats per row of x0
constexpr int kMT = (kM + 15) / 16;     // 7 tiles over detections (last is ragged)
constexpr int kNTiles = kHW / 16;       // 1600 tiles over pixels
constexpr int kNTPerWave = 8;           // N-tiles swept per wave (reuses A)
constexpr int kNGroups = kNTiles / kNTPerWave; // 200
constexpr float kScale = 0.25f;         // POOLER_SCALE

// flat output layout (tuple concatenated in return order, as float)
constexpr int kOffNumDet  = 0;
constexpr int kOffBoxes   = kOffNumDet + kB;           // 16
constexpr int kOffScores  = kOffBoxes + kB * kM * 4;   // 6416
constexpr int kOffClasses = kOffScores + kB * kM;      // 8016
constexpr int kOffMasks   = kOffClasses + kB * kM;     // 9616
} // namespace

// Small pass-through outputs: num_det, det_boxes, det_scores, det_classes.
__global__ void head_kernel(const int* __restrict__ num_det,
                            const float* __restrict__ det_boxes,
                            const float* __restrict__ det_scores,
                            const int* __restrict__ det_classes,
                            float* __restrict__ out) {
    const int i = blockIdx.x * blockDim.x + threadIdx.x;
    if (i < kB)           out[kOffNumDet  + i] = (float)num_det[i];
    if (i < kB * kM * 4)  out[kOffBoxes   + i] = det_boxes[i];
    if (i < kB * kM)      out[kOffScores  + i] = det_scores[i];
    if (i < kB * kM)      out[kOffClasses + i] = (float)det_classes[i];
}

// Sweep kNTPerWave pixel tiles for one (batch, m-tile) with a fixed A matrix.
// EDGE=false: hot path, unconditional stores, no divergence anywhere.
// EDGE=true : ragged last m-tile (detections 96..111), stores guarded.
template <bool EDGE>
__device__ __forceinline__ void sweep_tiles(
        const v16h& a, const float* __restrict__ pbase,
        const float* bx1, const float* by1, const float* bx2, const float* by2,
        float* __restrict__ orow0, int ng, int row, int k0, int hi, int mt) {
    #pragma unroll 2
    for (int nt = 0; nt < kNTPerWave; ++nt) {
        const int px = (ng * kNTPerWave + nt) * 16 + row; // pixel column (N index)

        // ---- B matrix: 32x16 f16 column of proto (L2-resident, regular TH) ----
        v16h bt;
        #pragma unroll
        for (int e = 0; e < 16; ++e)
            bt[e] = (_Float16)pbase[(size_t)(k0 + (e & 7) + ((e >> 3) << 4)) * kHW + px];

        v8f c = {};
        c = __builtin_amdgcn_wmma_f32_16x16x32_f16(
                /*neg_a=*/false, a, /*neg_b=*/false, bt,
                /*c_mod=*/(short)0, c, /*reuse_a=*/false, /*reuse_b=*/false);

        // ---- epilogue: fast sigmoid + branchless crop ----
        const float xf = (float)(px % kW);
        const float yf = (float)(px / kW);
        float res[8];
        #pragma unroll
        for (int r = 0; r < 8; ++r) {
            const float s = __builtin_amdgcn_rcpf(1.0f + __expf(-c[r]));
            const bool in = (xf >= bx1[r]) & (xf < bx2[r]) &
                            (yf >= by1[r]) & (yf < by2[r]);
            res[r] = in ? s : 0.0f;       // v_cndmask, no branches
        }

        // ---- stores: nontemporal (write-once stream; keep proto in L2) ----
        float* orow = orow0 + px;
        #pragma unroll
        for (int r = 0; r < 8; ++r) {
            if (!EDGE || (mt * 16 + hi * 8 + r < kM))
                __builtin_nontemporal_store(res[r], orow + (size_t)r * kHW);
        }
    }
}

// Fused gather -> WMMA GEMM (coefs x proto) -> sigmoid -> box crop -> store.
// One wave computes a 16(det) x 16(px) tile and sweeps kNTPerWave pixel tiles.
// Ragged edge (detections 100..111) handled by index clamping on the read side
// (garbage rows are simply never stored) so the hot path has zero divergence.
__global__ __launch_bounds__(256) void mask_wmma_kernel(
        const float* __restrict__ x0,
        const float* __restrict__ proto,
        const float* __restrict__ det_boxes,
        const int*   __restrict__ det_indices,
        float* __restrict__ out_masks) {
    const int lane = threadIdx.x & 31;
    const int wave = blockIdx.x * (blockDim.x >> 5) + (threadIdx.x >> 5);

    const int b   = wave / (kMT * kNGroups);
    const int rem = wave % (kMT * kNGroups);
    const int mt  = rem / kNGroups;       // detection tile (0..6), wave-uniform
    const int ng  = rem % kNGroups;       // pixel tile group (0..199)

    const int row = lane & 15;
    const int hi  = (lane >> 4) & 1;      // 0: lanes 0-15, 1: lanes 16-31

    // ---- A matrix: gathered mask coefficients, 16x32 f16 ----
    // lanes 0-15 (hi=0): elems 0..7 = K 0..7,  elems 8..15 = K 16..23
    // lanes 16-31(hi=1): elems 0..7 = K 8..15, elems 8..15 = K 24..31
    // Detection index clamped: rows >= kM compute garbage that is never stored.
    const int mA = min(mt * 16 + row, kM - 1);
    const int idx = det_indices[b * kM + mA];
    const float* srcA = x0 + ((size_t)b * kN + (size_t)idx) * kXS + (5 + kNC);
    const int kb = hi ? 8 : 0;
    v16h a;
    #pragma unroll
    for (int e = 0; e < 8; ++e)
        a[e] = (_Float16)__builtin_nontemporal_load(srcA + kb + e);
    #pragma unroll
    for (int e = 0; e < 8; ++e)
        a[8 + e] = (_Float16)__builtin_nontemporal_load(srcA + kb + 16 + e);

    // ---- per-lane crop boxes for my 8 accumulator rows (clamped, branchless) ----
    float bx1[8], by1[8], bx2[8], by2[8];
    #pragma unroll
    for (int r = 0; r < 8; ++r) {
        const int mr = min(mt * 16 + hi * 8 + r, kM - 1);
        const float4 bb = ((const float4*)det_boxes)[b * kM + mr];
        bx1[r] = bb.x * kScale; by1[r] = bb.y * kScale;
        bx2[r] = bb.z * kScale; by2[r] = bb.w * kScale;
    }

    const float* pbase = proto + (size_t)b * kNM * kHW;
    const int k0 = hi ? 8 : 0;            // B layout: lanes 0-15 K=0..7,16..23; 16-31 K=8..15,24..31
    float* orow0 = out_masks + ((size_t)(b * kM + mt * 16 + hi * 8)) * kHW;

    // Uniform branch hoisted OUTSIDE the sweep loop: hot path is divergence-free.
    if (mt != kMT - 1) {
        sweep_tiles<false>(a, pbase, bx1, by1, bx2, by2, orow0, ng, row, k0, hi, mt);
    } else {
        sweep_tiles<true >(a, pbase, bx1, by1, bx2, by2, orow0, ng, row, k0, hi, mt);
    }
}

extern "C" void kernel_launch(void* const* d_in, const int* in_sizes, int n_in,
                              void* d_out, int out_size, void* d_ws, size_t ws_size,
                              hipStream_t stream) {
    const float* x0          = (const float*)d_in[0];
    const float* proto       = (const float*)d_in[1];
    const int*   num_det     = (const int*)  d_in[2];
    const float* det_boxes   = (const float*)d_in[3];
    const float* det_scores  = (const float*)d_in[4];
    const int*   det_classes = (const int*)  d_in[5];
    const int*   det_indices = (const int*)  d_in[6];
    float* out = (float*)d_out;

    // Pass-through small outputs.
    const int headN = kB * kM * 4; // 6400 covers all four ranges
    head_kernel<<<(headN + 255) / 256, 256, 0, stream>>>(
        num_det, det_boxes, det_scores, det_classes, out);

    // Fused WMMA mask kernel: 16 batches x 7 m-tiles x 200 pixel-tile groups
    // = 22400 waves -> 2800 blocks of 256 threads (8 waves).
    const int totalWaves = kB * kMT * kNGroups;
    mask_wmma_kernel<<<totalWaves / 8, 256, 0, stream>>>(
        x0, proto, det_boxes, det_indices, out + kOffMasks);
}
